// slerp_59485297049913
// MI455X (gfx1250) — matrix operand order repaired and checked by reading, not compiled, not run on hardware
//
#include <hip/hip_runtime.h>
#include <hip/hip_bf16.h>

typedef __attribute__((ext_vector_type(16))) __bf16 v16bf;
typedef __attribute__((ext_vector_type(8)))  __bf16 v8bf;
typedef __attribute__((ext_vector_type(8)))  float  v8f;

#define BETA    5.0f
#define ALPHA   0.5f
#define TCOEF  (-0.75f)
#define DOT_THR 0.9995f

#define B_ROWS 8192
#define DDIM   1024
#define NCK    16000
#define CCOL   1000
#define CPAD   1008
#define NT_K   (NCK / 16)   // 1000 n-tiles of cache_keys
#define NT_V   (CPAD / 16)  // 63 n-tiles of values / clip cols

#define SMA_STRIDE 1032     // bf16 elements per res row in LDS (pad, 16B-aligned)
#define SMT_STRIDE 136      // bf16 elements per weights row in LDS

// ---------------------------------------------------------------------------
// Kernel 1: slerp(anchor, source, t=-0.75) + L2 normalize, emit bf16 res[B,D]
// ---------------------------------------------------------------------------
__global__ __launch_bounds__(256)
void slerp_norm_kernel(const float* __restrict__ xa,
                       const float* __restrict__ xs,
                       __bf16* __restrict__ resB) {
  __shared__ float red[256];
  const int row = blockIdx.x;
  const int t   = threadIdx.x;

  const float4 a = ((const float4*)(xa + (size_t)row * DDIM))[t];
  const float4 s = ((const float4*)(xs + (size_t)row * DDIM))[t];

  float p = a.x * s.x + a.y * s.y + a.z * s.z + a.w * s.w;
  red[t] = p;
  __syncthreads();
  for (int o = 128; o > 0; o >>= 1) {
    if (t < o) red[t] += red[t + o];
    __syncthreads();
  }
  const float dot = red[0];
  __syncthreads();

  float c0, c1;
  if (fabsf(dot) > DOT_THR) {           // lerp branch: a + T(s-a)
    c0 = 1.0f - TCOEF;
    c1 = TCOEF;
  } else {                               // slerp branch
    const float th0 = acosf(dot);
    const float ist = 1.0f / sinf(th0);
    c0 = sinf(th0 * (1.0f - TCOEF)) * ist;
    c1 = sinf(th0 * TCOEF) * ist;
  }

  float4 r;
  r.x = c0 * a.x + c1 * s.x;
  r.y = c0 * a.y + c1 * s.y;
  r.z = c0 * a.z + c1 * s.z;
  r.w = c0 * a.w + c1 * s.w;

  p = r.x * r.x + r.y * r.y + r.z * r.z + r.w * r.w;
  red[t] = p;
  __syncthreads();
  for (int o = 128; o > 0; o >>= 1) {
    if (t < o) red[t] += red[t + o];
    __syncthreads();
  }
  const float inv = rsqrtf(red[0]);

  __bf16* dst = resB + (size_t)row * DDIM + t * 4;
  dst[0] = (__bf16)(r.x * inv);
  dst[1] = (__bf16)(r.y * inv);
  dst[2] = (__bf16)(r.z * inv);
  dst[3] = (__bf16)(r.w * inv);
}

// ---------------------------------------------------------------------------
// Kernel 2: pack a f32 [Krows x Ncols] B-operand into bf16 WMMA B-fragment
// tiles (32x16). Tile (kt, nt) at dst + (kt*NT + nt)*512 elements; within a
// tile, lane L owns 32 contiguous bytes: element e of lane (n=L&15, g=L>>4)
// is B[kt*32 + (e&7) + 8*g + 16*(e>>3)][nt*16 + n]  (ISA 7.12.2 16-bit B layout).
// ---------------------------------------------------------------------------
__global__ __launch_bounds__(256)
void pack_b_kernel(const float* __restrict__ src, __bf16* __restrict__ dst,
                   int Ncols, int NT, float scale) {
  __shared__ float sm[32][129];
  const int t  = threadIdx.x;
  const int n0 = blockIdx.x * 128;
  const int k0 = blockIdx.y * 32;

#pragma unroll
  for (int i = 0; i < 16; ++i) {
    const int idx = t + i * 256;
    const int r = idx >> 7;
    const int c = idx & 127;
    const int n = n0 + c;
    sm[r][c] = (n < Ncols) ? src[(size_t)(k0 + r) * Ncols + n] * scale : 0.0f;
  }
  __syncthreads();

  const int w    = t >> 5;
  const int lane = t & 31;
  const int nl   = lane & 15;
  const int g    = lane >> 4;
  const int nt   = blockIdx.x * 8 + w;
  if (nt < NT) {
    v16bf v;
#pragma unroll
    for (int e = 0; e < 16; ++e) v[e] = (__bf16)sm[(e & 7) + 8 * g + 16 * (e >> 3)][w * 16 + nl];
    *((v16bf*)(dst + ((size_t)blockIdx.y * NT + nt) * 512 + lane * 16)) = v;
  }
}

// ---------------------------------------------------------------------------
// Kernel 3: fused  out = 100*res@W + alpha * exp(beta*(res@K - 1)) @ V
// Block: 32 rows of B (two M-tiles) x all 1008 (padded) columns, 8 waves.
// Each wave owns both M-tiles so every B fragment is reused by 2 WMMAs.
// ---------------------------------------------------------------------------
__global__ __launch_bounds__(256, 1)
void tip_fused_kernel(const __bf16* __restrict__ resB,
                      const __bf16* __restrict__ Kp,
                      const __bf16* __restrict__ Vp,
                      const __bf16* __restrict__ Wp,
                      float* __restrict__ out) {
  __shared__ __align__(16) __bf16 smA[32 * SMA_STRIDE];  // res rows (bf16)
  __shared__ __align__(16) __bf16 smT[32 * SMT_STRIDE];  // exp-weights [32x128]

  const int t    = threadIdx.x;
  const int row0 = blockIdx.x * 32;

  // ---- Stage res rows into LDS via CDNA5 async global->LDS copies
  // (tracked by ASYNCcnt, bypasses VGPRs). 32 rows x 1024 bf16 = 64KB.
  {
    const __bf16* gsrc = resB + (size_t)row0 * DDIM;
#pragma unroll
    for (int i = 0; i < 16; ++i) {
      const int c   = t + i * 256;                 // 4096 chunks of 8 bf16
      const int r   = c >> 7;
      const int off = (c & 127) * 8;
      unsigned lds_addr = (unsigned)(size_t)(&smA[r * SMA_STRIDE + off]);
      unsigned long long gaddr =
          (unsigned long long)(size_t)(gsrc + (size_t)r * DDIM + off);
      asm volatile("global_load_async_to_lds_b128 %0, %1, off"
                   :: "v"(lds_addr), "v"(gaddr) : "memory");
    }
    asm volatile("s_wait_asynccnt 0x0" ::: "memory");
  }
  __syncthreads();

  const int w    = t >> 5;
  const int lane = t & 31;
  const int ml   = lane & 15;
  const int g    = lane >> 4;

  const __bf16* smA0 = &smA[ml * SMA_STRIDE];            // M-tile 0 row
  const __bf16* smA1 = &smA[(16 + ml) * SMA_STRIDE];     // M-tile 1 row
  const __bf16* smT0 = &smT[ml * SMT_STRIDE];
  const __bf16* smT1 = &smT[(16 + ml) * SMT_STRIDE];

  v8f accC0[8] = {};   // M-tile 0, n-tiles nt = w + 8j
  v8f accC1[8] = {};   // M-tile 1

  for (int nc0 = 0; nc0 < NCK; nc0 += 128) {
    // ---- Stage A: affinity tiles [2 x 16x16] at n-tile ntK, K frag reused 2x
    const int ntK = (nc0 >> 4) + w;
    v8f acc0 = {};
    v8f acc1 = {};
#pragma unroll 4
    for (int kt = 0; kt < 32; ++kt) {
      const v8bf a0lo = *((const v8bf*)&smA0[kt * 32 + 8 * g]);
      const v8bf a0hi = *((const v8bf*)&smA0[kt * 32 + 16 + 8 * g]);
      const v8bf a1lo = *((const v8bf*)&smA1[kt * 32 + 8 * g]);
      const v8bf a1hi = *((const v8bf*)&smA1[kt * 32 + 16 + 8 * g]);
      v16bf A0, A1;
#pragma unroll
      for (int e = 0; e < 8; ++e) {
        A0[e] = a0lo[e]; A0[e + 8] = a0hi[e];
        A1[e] = a1lo[e]; A1[e + 8] = a1hi[e];
      }
      const v16bf Bf =
          *((const v16bf*)(Kp + ((size_t)kt * NT_K + ntK) * 512 + lane * 16));
      acc0 = __builtin_amdgcn_wmma_f32_16x16x32_bf16(false, A0, false, Bf,
                                                     (short)0, acc0, false, false);
      acc1 = __builtin_amdgcn_wmma_f32_16x16x32_bf16(false, A1, false, Bf,
                                                     (short)0, acc1, false, false);
    }
    // t = alpha * exp(beta*(aff-1)) -> bf16 into smT (D layout: row r+8g, col ml)
#pragma unroll
    for (int r = 0; r < 8; ++r) {
      const float t0 = ALPHA * __expf(BETA * (acc0[r] - 1.0f));
      const float t1 = ALPHA * __expf(BETA * (acc1[r] - 1.0f));
      smT[(r + 8 * g) * SMT_STRIDE + w * 16 + ml]        = (__bf16)t0;
      smT[(16 + r + 8 * g) * SMT_STRIDE + w * 16 + ml]   = (__bf16)t1;
    }
    __syncthreads();

    // ---- Stage B: weights @ V, each V frag reused for 2 M-tiles
#pragma unroll
    for (int kk = 0; kk < 4; ++kk) {
      const v8bf a0lo = *((const v8bf*)&smT0[kk * 32 + 8 * g]);
      const v8bf a0hi = *((const v8bf*)&smT0[kk * 32 + 16 + 8 * g]);
      const v8bf a1lo = *((const v8bf*)&smT1[kk * 32 + 8 * g]);
      const v8bf a1hi = *((const v8bf*)&smT1[kk * 32 + 16 + 8 * g]);
      v16bf A0, A1;
#pragma unroll
      for (int e = 0; e < 8; ++e) {
        A0[e] = a0lo[e]; A0[e + 8] = a0hi[e];
        A1[e] = a1lo[e]; A1[e + 8] = a1hi[e];
      }
      const int ktV = (nc0 >> 5) + kk;
#pragma unroll
      for (int j = 0; j < 8; ++j) {
        const int nt = w + 8 * j;
        if (nt >= NT_V) continue;
        const v16bf Bf =
            *((const v16bf*)(Vp + ((size_t)ktV * NT_V + nt) * 512 + lane * 16));
        accC0[j] = __builtin_amdgcn_wmma_f32_16x16x32_bf16(
            false, A0, false, Bf, (short)0, accC0[j], false, false);
        accC1[j] = __builtin_amdgcn_wmma_f32_16x16x32_bf16(
            false, A1, false, Bf, (short)0, accC1[j], false, false);
      }
    }
    __syncthreads();   // smT reused next chunk
  }

  // ---- clip logits: 100*res@W (100 folded into Wp), reuse res in LDS
#pragma unroll
  for (int j = 0; j < 8; ++j) {
    const int nt = w + 8 * j;
    if (nt >= NT_V) continue;
    v8f acc0 = {};
    v8f acc1 = {};
#pragma unroll 4
    for (int kt = 0; kt < 32; ++kt) {
      const v8bf a0lo = *((const v8bf*)&smA0[kt * 32 + 8 * g]);
      const v8bf a0hi = *((const v8bf*)&smA0[kt * 32 + 16 + 8 * g]);
      const v8bf a1lo = *((const v8bf*)&smA1[kt * 32 + 8 * g]);
      const v8bf a1hi = *((const v8bf*)&smA1[kt * 32 + 16 + 8 * g]);
      v16bf A0, A1;
#pragma unroll
      for (int e = 0; e < 8; ++e) {
        A0[e] = a0lo[e]; A0[e + 8] = a0hi[e];
        A1[e] = a1lo[e]; A1[e + 8] = a1hi[e];
      }
      const v16bf Bf =
          *((const v16bf*)(Wp + ((size_t)kt * NT_V + nt) * 512 + lane * 16));
      acc0 = __builtin_amdgcn_wmma_f32_16x16x32_bf16(false, A0, false, Bf,
                                                     (short)0, acc0, false, false);
      acc1 = __builtin_amdgcn_wmma_f32_16x16x32_bf16(false, A1, false, Bf,
                                                     (short)0, acc1, false, false);
    }
#pragma unroll
    for (int r = 0; r < 8; ++r) {
      accC0[j][r] += acc0[r];
      accC1[j][r] += acc1[r];
    }
  }

  // ---- store [32 x 1000] strip
#pragma unroll
  for (int j = 0; j < 8; ++j) {
    const int nt = w + 8 * j;
    if (nt >= NT_V) continue;
    const int c0 = nt * 16 + ml;
    if (c0 < CCOL) {
#pragma unroll
      for (int r = 0; r < 8; ++r) {
        out[(size_t)(row0 + r + 8 * g) * CCOL + c0]      = accC0[j][r];
        out[(size_t)(row0 + 16 + r + 8 * g) * CCOL + c0] = accC1[j][r];
      }
    }
  }
}

// ---------------------------------------------------------------------------
// Workspace layout (bytes):
//   resB : 8192*1024*2               = 16,777,216
//   Kp   : 32 kt * 1000 nt * 1024    = 32,768,000
//   Vp   : 500 kt * 63 nt * 1024     = 32,256,000
//   Wp   : 32 kt * 63 nt * 1024      =  2,064,384
//   total ~= 83.9 MB
// ---------------------------------------------------------------------------
extern "C" void kernel_launch(void* const* d_in, const int* in_sizes, int n_in,
                              void* d_out, int out_size, void* d_ws, size_t ws_size,
                              hipStream_t stream) {
  const float* xa = (const float*)d_in[0];   // x_anchor   [8192,1024]
  const float* xs = (const float*)d_in[1];   // x_source   [8192,1024]
  const float* ck = (const float*)d_in[2];   // cache_keys [1024,16000]
  const float* cv = (const float*)d_in[3];   // cache_vals [16000,1000]
  const float* cw = (const float*)d_in[4];   // clip_w     [1024,1000]
  float* out = (float*)d_out;

  char* ws = (char*)d_ws;
  __bf16* resB = (__bf16*)(ws);
  __bf16* Kp   = (__bf16*)(ws + 16777216);
  __bf16* Vp   = (__bf16*)(ws + 16777216 + 32768000);
  __bf16* Wp   = (__bf16*)(ws + 16777216 + 32768000 + 32256000);

  slerp_norm_kernel<<<B_ROWS, 256, 0, stream>>>(xa, xs, resB);
  pack_b_kernel<<<dim3(125, 32), 256, 0, stream>>>(ck, Kp, NCK, NT_K, 1.0f);
  pack_b_kernel<<<dim3(8, 500), 256, 0, stream>>>(cv, Vp, CCOL, NT_V, 1.0f);
  pack_b_kernel<<<dim3(8, 32),  256, 0, stream>>>(cw, Wp, CCOL, NT_V, 100.0f);
  tip_fused_kernel<<<B_ROWS / 32, 256, 0, stream>>>(resB, Kp, Vp, Wp, out);
}
